// MultiHeadAttention_40441412059740
// MI455X (gfx1250) — matrix operand order repaired and checked by reading, not compiled, not run on hardware
//
#include <hip/hip_runtime.h>
#include <hip/hip_bf16.h>

// -------------------------------------------------------------------------
// MHA forward for MI455X (gfx1250, wave32, WMMA 16x16x32 bf16).
// B=2, S=2048, E=1024, H=16, DK=64, ROT_DIM=32.
// d_out: x (B,S,E) f32 then attn (B,H,S,S) f32.
// d_ws:  Q,K (B,H,S,DK) bf16 ; V transposed (B,H,DK,S) bf16 (3 x 8 MB).
// -------------------------------------------------------------------------

typedef __bf16 bf16;
typedef __attribute__((ext_vector_type(16))) __bf16 v16bf;
typedef __attribute__((ext_vector_type(8)))  float  v8f;

#define B_   2
#define S_   2048
#define E_   1024
#define H_   16
#define DK_  64
#define ROT_ 32

// ISA 7.12.2: 16-bit A-matrix 16x32 — lane L holds row M=L&15.
__device__ __forceinline__ int kpairA(int v, int hi) {
    return 2 * v + (hi ? 16 : 0);
}
// 16-bit B-matrix 32x16 — lane L holds column N=L&15; K contiguous across v.
__device__ __forceinline__ int kpairB(int v, int hi) {
    return 2 * v + (hi ? 16 : 0);
}
__device__ __forceinline__ void put2(v16bf& f, int v, unsigned int bits) {
    union { unsigned int u; __bf16 h[2]; } cv;
    cv.u = bits;
    f[2 * v]     = cv.h[0];
    f[2 * v + 1] = cv.h[1];
}
__device__ __forceinline__ void wait_asynccnt0() {
#if __has_builtin(__builtin_amdgcn_s_wait_asynccnt)
    __builtin_amdgcn_s_wait_asynccnt(0);
#else
    asm volatile("s_wait_asynccnt 0x0" ::: "memory");
#endif
}
// Per-lane async copy of 16 bytes global->LDS (GVS mode, ASYNCcnt-tracked).
// lds_off: LDS byte address (low 32 bits of flat shared address);
// goff:    per-lane byte offset from base; base: uniform 64-bit address.
__device__ __forceinline__ void async_g2l_b128(unsigned lds_off, unsigned goff,
                                               unsigned long long base) {
    asm volatile("global_load_async_to_lds_b128 %0, %1, %2"
                 :: "v"(lds_off), "v"(goff), "s"(base) : "memory");
}

// ======================= Kernel 1: QKV projection ========================
// grid (E/64, M/64, 3), block 128 (4 waves). out = X @ W^T + b.
// Raw f32 64x32 tiles of X and W staged via async-to-LDS; f32->bf16 at
// fragment build (contiguous runs -> ds_load_b128 + pack converts).
__global__ __launch_bounds__(128) void qkv_proj_kernel(
    const float* __restrict__ q_in, const float* __restrict__ k_in,
    const float* __restrict__ v_in,
    const float* __restrict__ Wq, const float* __restrict__ bq,
    const float* __restrict__ Wk, const float* __restrict__ bk,
    const float* __restrict__ Wv, const float* __restrict__ bv,
    bf16* __restrict__ q_ws, bf16* __restrict__ k_ws, bf16* __restrict__ v_ws)
{
    __shared__ float Asf[64][32];   // 8 KB X tile (f32)
    __shared__ float Bsf[64][32];   // 8 KB W tile (f32)

    const int z = blockIdx.z;
    const float* X    = (z == 0) ? q_in : (z == 1) ? k_in : v_in;
    const float* W    = (z == 0) ? Wq   : (z == 1) ? Wk   : Wv;
    const float* bias = (z == 0) ? bq   : (z == 1) ? bk   : bv;
    bf16*        dst  = (z == 0) ? q_ws : (z == 1) ? k_ws : v_ws;

    const int n0 = blockIdx.x * 64;
    const int m0 = blockIdx.y * 64;
    const int t    = threadIdx.x;
    const int lane = t & 31;
    const int w    = t >> 5;
    const int hi   = (lane >= 16) ? 1 : 0;
    const int ll   = lane & 15;

    // staging geometry: each thread copies 4 rows x 16 bytes per buffer
    const int c4 = (t & 7) * 4;          // float column base (0,4,..,28)
    const int rr = t >> 3;               // row base 0..15
    const unsigned ldsA0 = (unsigned)(uintptr_t)&Asf[rr][c4];
    const unsigned ldsB0 = (unsigned)(uintptr_t)&Bsf[rr][c4];
    const unsigned long long Xb = (unsigned long long)(uintptr_t)X;
    const unsigned long long Wb = (unsigned long long)(uintptr_t)W;

    v8f acc0 = {}, acc1 = {}, acc2 = {}, acc3 = {};

    for (int kc = 0; kc < E_; kc += 32) {
#pragma unroll
        for (int i = 0; i < 4; ++i) {
            const int r = rr + 16 * i;
            async_g2l_b128(ldsA0 + i * 2048,
                           (unsigned)(((m0 + r) * E_ + kc + c4) * 4), Xb);
            async_g2l_b128(ldsB0 + i * 2048,
                           (unsigned)(((n0 + r) * E_ + kc + c4) * 4), Wb);
        }
        wait_asynccnt0();
        __syncthreads();

        v16bf a;
        const int mrow = 16 * w + ll;
#pragma unroll
        for (int v = 0; v < 8; ++v) {
            const int kk = kpairA(v, hi);
            a[2 * v]     = (bf16)Asf[mrow][kk];
            a[2 * v + 1] = (bf16)Asf[mrow][kk + 1];
        }
#pragma unroll
        for (int nt = 0; nt < 4; ++nt) {
            v16bf bm;
            const int n = nt * 16 + ll;
#pragma unroll
            for (int v = 0; v < 8; ++v) {
                const int kk = kpairB(v, hi);
                bm[2 * v]     = (bf16)Bsf[n][kk];
                bm[2 * v + 1] = (bf16)Bsf[n][kk + 1];
            }
            v8f& acc = (nt == 0) ? acc0 : (nt == 1) ? acc1 : (nt == 2) ? acc2 : acc3;
            acc = __builtin_amdgcn_wmma_f32_16x16x32_bf16(
                false, a, false, bm, (short)0, acc, false, false);
        }
        __syncthreads();
    }

    // epilogue: +bias; Q/K scatter to (B,H,S,DK); V scatter transposed (B,H,DK,S)
#pragma unroll
    for (int nt = 0; nt < 4; ++nt) {
        const v8f& acc = (nt == 0) ? acc0 : (nt == 1) ? acc1 : (nt == 2) ? acc2 : acc3;
        const int n  = n0 + nt * 16 + ll;
        const float bb = bias[n];
        const int hh = n >> 6;           // head
        const int d  = n & (DK_ - 1);    // dim in head
#pragma unroll
        for (int v = 0; v < 8; ++v) {
            const int m = m0 + 16 * w + v + hi * 8;   // row in (B*S)
            const int b = m >> 11;
            const int s = m & (S_ - 1);
            const size_t idx = (z == 2)
                ? ((size_t)((b * H_ + hh) * DK_ + d) * S_ + s)    // V^T
                : ((size_t)((b * H_ + hh) * S_ + s) * DK_ + d);   // Q,K
            dst[idx] = (bf16)(acc[v] + bb);
        }
    }
}

// ========================= Kernel 2: RoPE ================================
__global__ void rope_kernel(bf16* __restrict__ q_ws, bf16* __restrict__ k_ws)
{
    const int PP = B_ * H_ * S_ * (ROT_ / 2);
    int idx = blockIdx.x * blockDim.x + threadIdx.x;
    bf16* base = q_ws;
    if (idx >= PP) { idx -= PP; base = k_ws; }

    const int i  = idx & 15;
    const int s  = (idx >> 4) & (S_ - 1);
    const int bh = idx >> (4 + 11);

    const float ang = (float)s * __expf(-0.5756462732485115f * (float)i);
    float sn, cs;
    __sincosf(ang, &sn, &cs);

    const size_t off = ((size_t)bh * S_ + s) * DK_ + 2 * i;
    const float e = (float)base[off];
    const float o = (float)base[off + 1];
    base[off]     = (bf16)(e * cs - o * sn);
    base[off + 1] = (bf16)(o * cs + e * sn);
}

// ==================== Kernel 3: fused attention ==========================
// grid (S/16, B*H), block 128 (4 waves), 128 KB dynamic LDS score stripe.
__global__ __launch_bounds__(128) void attn_kernel(
    const bf16* __restrict__ q_ws, const bf16* __restrict__ k_ws,
    const bf16* __restrict__ v_ws,   // V^T: (B,H,DK,S)
    float* __restrict__ x_out, float* __restrict__ attn_out)
{
    extern __shared__ float sc[];   // [16][S_] f32

    const int q0 = blockIdx.x * 16;
    const int bh = blockIdx.y;
    const int b  = bh / H_;
    const int h  = bh % H_;
    const int t    = threadIdx.x;
    const int lane = t & 31;
    const int w    = t >> 5;
    const int hi   = (lane >= 16) ? 1 : 0;
    const int ll   = lane & 15;

    const bf16* qslab = q_ws + (size_t)bh * S_ * DK_;
    const bf16* kslab = k_ws + (size_t)bh * S_ * DK_;

    // Q A-fragments (K-dims 0..31, 32..63), held in VGPRs for whole kernel
    v16bf a0, a1;
    {
        const int m = q0 + ll;
#pragma unroll
        for (int v = 0; v < 8; ++v) {
            const int kk = kpairA(v, hi);
            put2(a0, v, *(const unsigned int*)&qslab[(size_t)m * DK_ + kk]);
            put2(a1, v, *(const unsigned int*)&qslab[(size_t)m * DK_ + 32 + kk]);
        }
    }

    // ---- scores: 128 key-tiles of 16, interleaved over 4 waves ----
    for (int kt = w; kt < S_ / 16; kt += 4) {
        const int nkey = kt * 16 + ll;
        __builtin_prefetch(&kslab[(size_t)(nkey + 64) * DK_], 0, 3); // next tile
        v16bf b0, b1;
#pragma unroll
        for (int v = 0; v < 8; ++v) {
            const int kk = kpairB(v, hi);   // contiguous across v -> b128 loads
            put2(b0, v, *(const unsigned int*)&kslab[(size_t)nkey * DK_ + kk]);
            put2(b1, v, *(const unsigned int*)&kslab[(size_t)nkey * DK_ + 32 + kk]);
        }
        v8f acc = {};
        acc = __builtin_amdgcn_wmma_f32_16x16x32_bf16(false, a0, false, b0, (short)0, acc, false, false);
        acc = __builtin_amdgcn_wmma_f32_16x16x32_bf16(false, a1, false, b1, (short)0, acc, false, false);
#pragma unroll
        for (int v = 0; v < 8; ++v)
            sc[(v + hi * 8) * S_ + kt * 16 + ll] = acc[v] * 0.125f;  // 1/sqrt(64)
    }
    __syncthreads();

    // ---- softmax: 4 rows per wave, shfl-xor wave32 reductions ----
    float* attn_base = attn_out + ((size_t)bh * S_ + q0) * S_;
    for (int r = w * 4; r < w * 4 + 4; ++r) {
        float* row = sc + r * S_;
        float mx = -3.4e38f;
        for (int j = lane; j < S_; j += 32) mx = fmaxf(mx, row[j]);
#pragma unroll
        for (int off = 16; off; off >>= 1) mx = fmaxf(mx, __shfl_xor(mx, off, 32));
        float sum = 0.f;
        for (int j = lane; j < S_; j += 32) {
            float e = __expf(row[j] - mx);
            row[j] = e;
            sum += e;
        }
#pragma unroll
        for (int off = 16; off; off >>= 1) sum += __shfl_xor(sum, off, 32);
        const float inv = 1.0f / sum;
        float* arow = attn_base + (size_t)r * S_;
        for (int j = lane; j < S_; j += 32) {
            const float p = row[j] * inv;
            row[j]  = p;    // WMMA A operand below
            arow[j] = p;    // the single 537 MB attn write
        }
    }
    __syncthreads();

    // ---- x = attn @ V^T-layout: wave w owns one 16-wide N-tile ----
    v8f acc = {};
    const int n = w * 16 + ll;
    const bf16* vrow = v_ws + ((size_t)bh * DK_ + n) * S_;  // V^T row: dim n, all keys
    for (int kc = 0; kc < S_; kc += 32) {
        v16bf a, bb;
#pragma unroll
        for (int v = 0; v < 8; ++v) {
            const int kk = kc + kpairA(v, hi);
            a[2 * v]     = (bf16)sc[ll * S_ + kk];
            a[2 * v + 1] = (bf16)sc[ll * S_ + kk + 1];
            // contiguous 32B run across v -> global_load_b128 pairs
            put2(bb, v, *(const unsigned int*)&vrow[kc + kpairB(v, hi)]);
        }
        acc = __builtin_amdgcn_wmma_f32_16x16x32_bf16(false, a, false, bb, (short)0, acc, false, false);
    }
#pragma unroll
    for (int v = 0; v < 8; ++v) {
        const int srow = q0 + v + hi * 8;
        x_out[(size_t)(b * S_ + srow) * E_ + h * DK_ + n] = acc[v];
    }
}

// ============================ launch =====================================
extern "C" void kernel_launch(void* const* d_in, const int* in_sizes, int n_in,
                              void* d_out, int out_size, void* d_ws, size_t ws_size,
                              hipStream_t stream)
{
    (void)in_sizes; (void)n_in; (void)out_size; (void)ws_size;

    const float* q_in = (const float*)d_in[0];
    const float* k_in = (const float*)d_in[1];
    const float* v_in = (const float*)d_in[2];
    const float* Wq   = (const float*)d_in[3];
    const float* bq   = (const float*)d_in[4];
    const float* Wk   = (const float*)d_in[5];
    const float* bk   = (const float*)d_in[6];
    const float* Wv   = (const float*)d_in[7];
    const float* bv   = (const float*)d_in[8];

    const size_t slab = (size_t)B_ * H_ * S_ * DK_;
    bf16* q_ws = (bf16*)d_ws;
    bf16* k_ws = q_ws + slab;
    bf16* v_ws = k_ws + slab;   // transposed (B,H,DK,S)

    float* x_out    = (float*)d_out;
    float* attn_out = x_out + (size_t)B_ * S_ * E_;

    const int smem_bytes = 16 * S_ * (int)sizeof(float);   // 128 KB
    hipFuncSetAttribute(reinterpret_cast<const void*>(attn_kernel),
                        hipFuncAttributeMaxDynamicSharedMemorySize, smem_bytes);

    dim3 g1(E_ / 64, (B_ * S_) / 64, 3);
    qkv_proj_kernel<<<g1, 128, 0, stream>>>(q_in, k_in, v_in, Wq, bq, Wk, bk,
                                            Wv, bv, q_ws, k_ws, v_ws);

    const int total_pairs = 2 * B_ * H_ * S_ * (ROT_ / 2);
    rope_kernel<<<total_pairs / 256, 256, 0, stream>>>(q_ws, k_ws);

    dim3 g3(S_ / 16, B_ * H_);
    attn_kernel<<<g3, 128, smem_bytes, stream>>>(q_ws, k_ws, v_ws, x_out, attn_out);
}
